// BaseDependentAttentionLayer_68350109549064
// MI455X (gfx1250) — matrix operand order repaired and checked by reading, not compiled, not run on hardware
//
#include <hip/hip_runtime.h>
#include <math.h>

#define D  64
#define H  4
#define HD 16
#define XT_STRIDE 68   // 16-row LDS tile row stride (floats); 68%64 spreads banks

typedef __attribute__((ext_vector_type(16))) _Float16 v16h;
typedef __attribute__((ext_vector_type(8)))  float    v8f;
typedef int v4i __attribute__((vector_size(16)));   // matches builtin param type

#define AS1 __attribute__((address_space(1)))
#define AS3 __attribute__((address_space(3)))

#if defined(__has_builtin)
#if __has_builtin(__builtin_amdgcn_global_load_async_to_lds_b128) && \
    __has_builtin(__builtin_amdgcn_s_wait_asynccnt)
#define GA_ASYNC_LDS 1
#endif
#endif

// ---------------------------------------------------------------------------
// Stage 8 consecutive floats (32B) from global into LDS.
// Async path: 2 x global_load_async_to_lds_b128 (ASYNCcnt tracked).
// ---------------------------------------------------------------------------
__device__ inline void copy_row8_to_lds(const float* gsrc, float* ldst) {
#ifdef GA_ASYNC_LDS
  AS1 v4i* g = (AS1 v4i*)(uintptr_t)gsrc;
  AS3 v4i* l = (AS3 v4i*)(unsigned int)(uintptr_t)ldst;  // low 32b = LDS offset
  __builtin_amdgcn_global_load_async_to_lds_b128(g, l, 0, 0);
  __builtin_amdgcn_global_load_async_to_lds_b128(g, l, 16, 0);
#else
  const float4* s = (const float4*)gsrc;
  float4*       d = (float4*)ldst;
  d[0] = s[0];
  d[1] = s[1];
#endif
}

__device__ inline void wait_lds_copies() {
#ifdef GA_ASYNC_LDS
  __builtin_amdgcn_s_wait_asynccnt(0);
#endif
  __syncthreads();
}

__device__ inline void pack4(v16h& dst, int base, float4 f) {
  dst[base + 0] = (_Float16)f.x;
  dst[base + 1] = (_Float16)f.y;
  dst[base + 2] = (_Float16)f.z;
  dst[base + 3] = (_Float16)f.w;
}

// ---------------------------------------------------------------------------
// A fragment (16x32 MxK, 16-bit layout §7.12.2) read from the LDS x-tile.
// Lane's halves are 16 contiguous floats: k = kBase+16*lh .. +15
//  ->  4 x ds_load_b128.
// ---------------------------------------------------------------------------
__device__ inline v16h load_A_frag_lds(const float* tile, int kBase, int lane) {
  const int m  = lane & 15;
  const int lh = lane >> 4;
  const float* rp = tile + m * XT_STRIDE + kBase + 16 * lh;
  const float4 f0 = ((const float4*)rp)[0];
  const float4 f1 = ((const float4*)rp)[1];
  const float4 f2 = ((const float4*)rp)[2];
  const float4 f3 = ((const float4*)rp)[3];
  v16h a;
  pack4(a, 0, f0); pack4(a, 4, f1); pack4(a, 8, f2); pack4(a, 12, f3);
  return a;
}

// ---------------------------------------------------------------------------
// B fragment (32x16 KxN): B[k][n] = W[n][k] (out = X @ W^T). Lane's halves are
// 16 contiguous floats of W row n starting at kBase+16*lh -> 4 x b128 loads.
// ---------------------------------------------------------------------------
__device__ inline v16h load_B_frag(const float* __restrict__ W, int nBase,
                                   int kBase, int lane) {
  const int n  = nBase + (lane & 15);
  const int lh = lane >> 4;
  const float4* p = (const float4*)(W + (size_t)n * D + kBase + 16 * lh);
  const float4 f0 = p[0], f1 = p[1], f2 = p[2], f3 = p[3];
  v16h b;
  pack4(b, 0, f0); pack4(b, 4, f1); pack4(b, 8, f2); pack4(b, 12, f3);
  return b;
}

__device__ inline v8f wmma_f16(v16h a, v16h b, v8f c) {
  return __builtin_amdgcn_wmma_f32_16x16x32_f16(false, a, false, b,
                                                (short)0, c, false, false);
}

__device__ inline void atomicMaxF(float* addr, float val) {
  // int-max for non-negative, uint-min for negative; init must be -inf.
  if (val >= 0.0f) atomicMax((int*)addr, __float_as_int(val));
  else             atomicMin((unsigned int*)addr, __float_as_uint(val));
}

__device__ inline void atomAddF(float* addr, float val) {
  __hip_atomic_fetch_add(addr, val, __ATOMIC_RELAXED, __HIP_MEMORY_SCOPE_AGENT);
}

// ---------------------------------------------------------------------------
// init: vals=0, m=-inf, z=0 (re-run every launch -> graph-replay safe)
// ---------------------------------------------------------------------------
__global__ __launch_bounds__(256) void ga_init_kernel(float* __restrict__ vals,
                                                      float* __restrict__ m,
                                                      float* __restrict__ z,
                                                      long long nd, long long nh) {
  long long idx = (long long)blockIdx.x * blockDim.x + threadIdx.x;
  if (idx < nd) vals[idx] = 0.0f;
  if (idx < nh) { m[idx] = -INFINITY; z[idx] = 0.0f; }
}

// ---------------------------------------------------------------------------
// Q/K/V GEMM. Block = 16 node rows, 4 waves; wave w owns column tile w.
// x tile async-staged to LDS; outputs staged to LDS for coalesced b128 stores.
// ---------------------------------------------------------------------------
__global__ __launch_bounds__(128) void ga_qkv_kernel(
    const float* __restrict__ x,
    const float* __restrict__ Wq, const float* __restrict__ bq,
    const float* __restrict__ Wk, const float* __restrict__ bk,
    const float* __restrict__ Wv, const float* __restrict__ bv,
    float* __restrict__ q, float* __restrict__ k, float* __restrict__ v,
    int N) {
  __shared__ float xtile[16 * XT_STRIDE];
  __shared__ float stile[16 * XT_STRIDE];

  const int lane    = threadIdx.x & 31;
  const int wave    = threadIdx.x >> 5;
  const int rowBase = blockIdx.x * 16;
  const int nBase   = wave * 16;

  // stage x[rowBase .. rowBase+15][0..63] into LDS (async, 32B per thread)
  {
    const int r   = threadIdx.x >> 3;     // 0..15
    const int seg = threadIdx.x & 7;      // 0..7 (8 floats each)
    int grow = rowBase + r;
    if (grow >= N) grow = N - 1;
    copy_row8_to_lds(x + (size_t)grow * D + seg * 8,
                     xtile + r * XT_STRIDE + seg * 8);
  }
  wait_lds_copies();

  const v16h a0 = load_A_frag_lds(xtile, 0,  lane);
  const v16h a1 = load_A_frag_lds(xtile, 32, lane);

  const int col  = nBase + (lane & 15);
  const int rOff = (lane >> 4) * 8;

  const float* Ws[3] = {Wq, Wk, Wv};
  const float* bs[3] = {bq, bk, bv};
  float*       os[3] = {q, k, v};

#pragma unroll
  for (int mtx = 0; mtx < 3; ++mtx) {
    const v16h b0 = load_B_frag(Ws[mtx], nBase, 0,  lane);
    const v16h b1 = load_B_frag(Ws[mtx], nBase, 32, lane);
    v8f c = {};
    c = wmma_f16(a0, b0, c);
    c = wmma_f16(a1, b1, c);
    const float bias = bs[mtx][col];
#pragma unroll
    for (int r = 0; r < 8; ++r)                 // C/D: M = r + 8*laneHi
      stile[(rOff + r) * XT_STRIDE + col] = c[r] + bias;
    __syncthreads();
    {
      const int r    = threadIdx.x >> 3;
      const int seg  = threadIdx.x & 7;
      const int grow = rowBase + r;
      if (grow < N) {
        const float4* sp = (const float4*)(stile + r * XT_STRIDE + seg * 8);
        float4*       dp = (float4*)(os[mtx] + (size_t)grow * D + seg * 8);
        dp[0] = sp[0];
        dp[1] = sp[1];
      }
    }
    __syncthreads();
  }
}

// ---------------------------------------------------------------------------
// Per-(edge,head) score + running segment max.  Gathers hit the 192MB L2.
// ---------------------------------------------------------------------------
__global__ __launch_bounds__(256) void ga_score_kernel(
    const float* __restrict__ q, const float* __restrict__ k,
    const int* __restrict__ ei, const float* __restrict__ ew,
    float* __restrict__ sc, float* __restrict__ m, int E) {
  long long idx = (long long)blockIdx.x * blockDim.x + threadIdx.x;
  if (idx >= (long long)E * H) return;
  const int e = (int)(idx >> 2);
  const int h = (int)(idx & 3);
  const int o = ei[e];
  const int d = ei[E + e];
  const float4* qp = (const float4*)(q + (size_t)o * D + h * HD);
  const float4* kp = (const float4*)(k + (size_t)d * D + h * HD);
  float s = 0.0f;
#pragma unroll
  for (int i = 0; i < 4; ++i) {
    const float4 a = qp[i], b = kp[i];
    s += a.x * b.x + a.y * b.y + a.z * b.z + a.w * b.w;
  }
  s = s * 0.25f * ew[(size_t)e * H + h];        // scale = HD^-0.5 = 0.25
  sc[idx] = s;
  atomicMaxF(m + (size_t)o * H + h, s);
}

// ---------------------------------------------------------------------------
// e = exp(s - m[o]); z[o] += e
// ---------------------------------------------------------------------------
__global__ __launch_bounds__(256) void ga_exp_kernel(
    float* __restrict__ sc, const float* __restrict__ m,
    float* __restrict__ z, const int* __restrict__ ei, int E) {
  long long idx = (long long)blockIdx.x * blockDim.x + threadIdx.x;
  if (idx >= (long long)E * H) return;
  const int e = (int)(idx >> 2);
  const int h = (int)(idx & 3);
  const int o = ei[e];
  const float val = expf(sc[idx] - m[(size_t)o * H + h]);
  sc[idx] = val;
  atomAddF(z + (size_t)o * H + h, val);
}

// ---------------------------------------------------------------------------
// attn = e/z; vals[o] += attn * v[d]  (16 global_atomic_add_f32 per thread)
// ---------------------------------------------------------------------------
__global__ __launch_bounds__(256) void ga_scatter_kernel(
    const float* __restrict__ sc, const float* __restrict__ z,
    const float* __restrict__ v, const int* __restrict__ ei,
    float* __restrict__ vals, int E) {
  long long idx = (long long)blockIdx.x * blockDim.x + threadIdx.x;
  if (idx >= (long long)E * H) return;
  const int e = (int)(idx >> 2);
  const int h = (int)(idx & 3);
  const int o = ei[e];
  const int d = ei[E + e];
  const float attn = sc[idx] / (z[(size_t)o * H + h] + 1e-16f);
  const float4* vp = (const float4*)(v + (size_t)d * D + h * HD);
  float* dst = vals + (size_t)o * D + h * HD;
#pragma unroll
  for (int i = 0; i < 4; ++i) {
    const float4 t = vp[i];
    atomAddF(dst + 4 * i + 0, attn * t.x);
    atomAddF(dst + 4 * i + 1, attn * t.y);
    atomAddF(dst + 4 * i + 2, attn * t.z);
    atomAddF(dst + 4 * i + 3, attn * t.w);
  }
}

// ---------------------------------------------------------------------------
// out = LN(vals @ Wo^T + bo) + x.  Async LDS stage -> WMMA -> fused LayerNorm.
// ---------------------------------------------------------------------------
__global__ __launch_bounds__(128) void ga_out_kernel(
    const float* __restrict__ vals, const float* __restrict__ Wo,
    const float* __restrict__ bo, const float* __restrict__ x,
    const float* __restrict__ gamma, const float* __restrict__ beta,
    float* __restrict__ out, int N) {
  __shared__ float vtile[16 * XT_STRIDE];
  __shared__ float otile[16 * XT_STRIDE];

  const int lane    = threadIdx.x & 31;
  const int wave    = threadIdx.x >> 5;
  const int rowBase = blockIdx.x * 16;
  const int nBase   = wave * 16;

  {
    const int r   = threadIdx.x >> 3;
    const int seg = threadIdx.x & 7;
    int grow = rowBase + r;
    if (grow >= N) grow = N - 1;
    copy_row8_to_lds(vals + (size_t)grow * D + seg * 8,
                     vtile + r * XT_STRIDE + seg * 8);
  }
  wait_lds_copies();

  const v16h a0 = load_A_frag_lds(vtile, 0,  lane);
  const v16h a1 = load_A_frag_lds(vtile, 32, lane);
  const v16h b0 = load_B_frag(Wo, nBase, 0,  lane);
  const v16h b1 = load_B_frag(Wo, nBase, 32, lane);
  v8f c = {};
  c = wmma_f16(a0, b0, c);
  c = wmma_f16(a1, b1, c);

  const int   col  = nBase + (lane & 15);
  const int   rOff = (lane >> 4) * 8;
  const float bias = bo[col];
#pragma unroll
  for (int r = 0; r < 8; ++r) otile[(rOff + r) * XT_STRIDE + col] = c[r] + bias;
  __syncthreads();

  // LayerNorm + residual: 8 threads per row, xor-shuffle reduce in 8-groups.
  const int row = threadIdx.x >> 3;             // 0..15
  const int sub = threadIdx.x & 7;              // 0..7
  const float* rp = otile + row * XT_STRIDE;
  float s = 0.0f, s2 = 0.0f;
#pragma unroll
  for (int j = 0; j < 8; ++j) {
    const float t = rp[sub * 8 + j];
    s += t; s2 += t * t;
  }
#pragma unroll
  for (int off = 1; off < 8; off <<= 1) {
    s  += __shfl_xor(s,  off, 8);
    s2 += __shfl_xor(s2, off, 8);
  }
  const float mu  = s * (1.0f / D);
  const float var = s2 * (1.0f / D) - mu * mu;
  const float rs  = rsqrtf(var + 1e-5f);

  const int grow = rowBase + row;
  if (grow < N) {
    const float4* xg = (const float4*)(x + (size_t)grow * D + sub * 8);
    const float4* gg = (const float4*)(gamma + sub * 8);
    const float4* bg = (const float4*)(beta + sub * 8);
    float4*       op = (float4*)(out + (size_t)grow * D + sub * 8);
#pragma unroll
    for (int i = 0; i < 2; ++i) {
      const float4 t = ((const float4*)(rp + sub * 8))[i];
      const float4 xv = xg[i], gv = gg[i], bv = bg[i];
      float4 r;
      r.x = xv.x + (t.x - mu) * rs * gv.x + bv.x;
      r.y = xv.y + (t.y - mu) * rs * gv.y + bv.y;
      r.z = xv.z + (t.z - mu) * rs * gv.z + bv.z;
      r.w = xv.w + (t.w - mu) * rs * gv.w + bv.w;
      op[i] = r;
    }
  }
}

// ---------------------------------------------------------------------------
extern "C" void kernel_launch(void* const* d_in, const int* in_sizes, int n_in,
                              void* d_out, int out_size, void* d_ws,
                              size_t ws_size, hipStream_t stream) {
  const float* x     = (const float*)d_in[0];
  const int*   ei    = (const int*)  d_in[1];   // edge_index [2,E]
  const float* ew    = (const float*)d_in[2];   // edge_weights [E,H]
  const float* Wq    = (const float*)d_in[3];
  const float* bq    = (const float*)d_in[4];
  const float* Wk    = (const float*)d_in[5];
  const float* bk    = (const float*)d_in[6];
  const float* Wv    = (const float*)d_in[7];
  const float* bv    = (const float*)d_in[8];
  const float* Wo    = (const float*)d_in[9];
  const float* bo    = (const float*)d_in[10];
  const float* gamma = (const float*)d_in[11];
  const float* beta  = (const float*)d_in[12];
  float* out = (float*)d_out;

  const int N = in_sizes[0] / D;
  const int E = in_sizes[1] / 2;

  float*       ws = (float*)d_ws;
  const size_t nd = (size_t)N * D;
  float* q    = ws;
  float* k    = q  + nd;
  float* v    = k  + nd;
  float* sc   = v  + nd;                       // E*H
  float* m    = sc + (size_t)E * H;            // N*H
  float* z    = m  + (size_t)N * H;            // N*H
  float* vals = z  + (size_t)N * H;            // N*D

  const int       tb16 = (N + 15) / 16;
  const long long EH   = (long long)E * H;
  const int       ebl  = (int)((EH + 255) / 256);

  ga_init_kernel<<<(int)((nd + 255) / 256), 256, 0, stream>>>(
      vals, m, z, (long long)nd, (long long)N * H);
  ga_qkv_kernel<<<tb16, 128, 0, stream>>>(x, Wq, bq, Wk, bk, Wv, bv, q, k, v, N);
  ga_score_kernel<<<ebl, 256, 0, stream>>>(q, k, ei, ew, sc, m, E);
  ga_exp_kernel<<<ebl, 256, 0, stream>>>(sc, m, z, ei, E);
  ga_scatter_kernel<<<ebl, 256, 0, stream>>>(sc, z, v, ei, vals, E);
  ga_out_kernel<<<tb16, 128, 0, stream>>>(vals, Wo, bo, x, gamma, beta, out, N);
}
